// HieformerSelfAttention_bilevel_31576599561168
// MI455X (gfx1250) — matrix operand order repaired and checked by reading, not compiled, not run on hardware
//
#include <hip/hip_runtime.h>

// ---------------------------------------------------------------------------
// Problem constants (match the JAX reference)
// ---------------------------------------------------------------------------
constexpr int kB = 2;
constexpr int kS = 4096;
constexpr int kE = 768;
constexpr int kH = 12;
constexpr int kD = 64;
constexpr int kW = 256;              // half window
// Padded key length: attention reads keys in [q0-256, q0+287] (17 blocks of
// 32 starting at q0-256). Padded index pk = key + kW ranges [0, 4623].
constexpr int kSP = kS + 2 * kW + 16;   // 4624

// ---------------------------------------------------------------------------
// WMMA fragment types (CDNA5: wave32, V_WMMA_F32_16X16X32_BF16)
// ---------------------------------------------------------------------------
typedef __attribute__((ext_vector_type(16))) __bf16 v16bf;
typedef __attribute__((ext_vector_type(8)))  __bf16 v8bf;
typedef __attribute__((ext_vector_type(8)))  float  v8f;

__device__ inline v16bf make_v16(v8bf lo, v8bf hi) {
  v16bf r;
#pragma unroll
  for (int i = 0; i < 8; ++i) { r[i] = lo[i]; r[i + 8] = hi[i]; }
  return r;
}

// A fragment: 16x32 bf16, source row-major (rows x K), leading dim ld.
// ISA layout: lane<16 -> row=lane, K = {0..7, 16..23}; lane>=16 -> row=lane-16,
// K = {8..15, 24..31}.  Both K-groups are 8 contiguous elements -> 2x b128.
__device__ inline v16bf load_a_frag(const __bf16* base, int ld, int row0,
                                    int k0, int lane) {
  const __bf16* p = base + (size_t)(row0 + (lane & 15)) * ld + k0 +
                    ((lane >> 4) << 3);
  v8bf lo = *(const v8bf*)(p);
  v8bf hi = *(const v8bf*)(p + 16);
  return make_v16(lo, hi);
}

// B fragment: 32x16 where B[k][n] = src[n0+n][k0+k], src row-major with
// leading dim ld.  ISA layout: lane = column n (mod 16); lanes 0-15 hold
// K = {0..7, 16..23}, lanes 16-31 hold K = {8..15, 24..31} (matches A).
__device__ inline v16bf load_b_frag(const __bf16* base, int ld, int n0,
                                    int k0, int lane) {
  const __bf16* p = base + (size_t)(n0 + (lane & 15)) * ld + k0 +
                    ((lane >> 4) << 3);
  v8bf lo = *(const v8bf*)(p);
  v8bf hi = *(const v8bf*)(p + 16);
  return make_v16(lo, hi);
}

__device__ inline v8f wmma_bf16(v16bf a, v16bf b, v8f c) {
  // (neg_a, A, neg_b, B, c_mod, C, reuse_a, reuse_b)
  return __builtin_amdgcn_wmma_f32_16x16x32_bf16(false, a, false, b,
                                                 (short)0, c, false, false);
}

// Reductions across a 16-lane half of the wave (C-matrix halves never mix
// lanes 0-15 with 16-31, so xor masks 1..8 are exactly right).
__device__ inline float half_max(float v) {
#pragma unroll
  for (int m = 1; m <= 8; m <<= 1) v = fmaxf(v, __shfl_xor(v, m, 32));
  return v;
}
__device__ inline float half_sum(float v) {
#pragma unroll
  for (int m = 1; m <= 8; m <<= 1) v += __shfl_xor(v, m, 32);
  return v;
}

// Intra-wave LDS write->read ordering (P staging buffers are wave-private).
__device__ inline void wave_lds_fence() {
  __builtin_amdgcn_fence(__ATOMIC_ACQ_REL, "wavefront");
  __builtin_amdgcn_wave_barrier();
}

// ---------------------------------------------------------------------------
// CDNA5 async global -> LDS copy (ASYNCcnt-tracked).  One instruction per
// wave; 16 bytes per lane.  lds_off is the LDS byte offset (low 32 bits of a
// generic pointer to __shared__ memory).
// ---------------------------------------------------------------------------
__device__ inline void async_load_b128(unsigned lds_off, const void* gptr) {
  asm volatile("global_load_async_to_lds_b128 %0, %1, off"
               :
               : "v"(lds_off), "v"((unsigned long long)(uintptr_t)gptr)
               : "memory");
}
__device__ inline void wait_async1() {
  asm volatile("s_wait_asynccnt 0x1" ::: "memory");
}
__device__ inline void wait_async0() {
  asm volatile("s_wait_asynccnt 0x0" ::: "memory");
}

// ---------------------------------------------------------------------------
// Kernel 1: fp32 -> bf16 conversion
// ---------------------------------------------------------------------------
__global__ void f32_to_bf16_kernel(const float* __restrict__ in,
                                   __bf16* __restrict__ out, int n) {
  int i = blockIdx.x * blockDim.x + threadIdx.x;
  if (i < n) out[i] = (__bf16)in[i];
}

// ---------------------------------------------------------------------------
// Kernel 2: fused QKV projection  out = hs @ W^T + b, split heads.
//   z==0: Q scaled by 1/sqrt(D) -> qbuf (B,H,S,D)
//   z==1: K                     -> kbuf (B,H,SP,D)   (+kW key offset)
//   z==2: V transposed          -> vtbuf(B,H,D,SP)   (+kW key offset)
// Block = 256 threads = 8 waves; each wave owns a 32x64 output tile.
// The shared 64x32 weight tile is staged to LDS with TRIPLE-buffered
// global_load_async_to_lds_b128: exactly one s_wait_asynccnt + one barrier
// per chunk (issue of chunk j+2 reuses the buffer read at step j-1, which is
// protected by step j's barrier).  Straight-line 3-step body keeps fixed LDS
// offsets and long barrier-free WMMA runs.
// ---------------------------------------------------------------------------
__global__ __launch_bounds__(256) void qkv_gemm_kernel(
    const __bf16* __restrict__ hsb, const __bf16* __restrict__ wall,
    const float* __restrict__ bq, const float* __restrict__ bk,
    const float* __restrict__ bv, __bf16* __restrict__ qbuf,
    __bf16* __restrict__ kbuf, __bf16* __restrict__ vtbuf) {
  __shared__ __align__(16) __bf16 bsh[3][64][32];  // triple-buffered B tile

  const int tid = threadIdx.x;
  const int lane = tid & 31;
  const int wave = tid >> 5;
  const int z = blockIdx.z;
  const int row0 = blockIdx.x * 256 + wave * 32;  // 0..8191, 32 rows/wave
  const int n0 = blockIdx.y * 64;                 // 0..704
  const __bf16* Wz = wall + (size_t)z * kE * kE;  // row-major (out_dim, in_dim)
  const float* bias = (z == 0) ? bq : (z == 1) ? bk : bv;

  // Cooperative async stage of one 64x32 B tile: 256 lanes x 16B = 4 KB.
  const int srow = tid >> 2;  // 0..63 (output-feature row of W)
  const int schk = tid & 3;   // 0..3  (16-byte chunk within the 32-k row)
  const __bf16* sbase = Wz + (size_t)(n0 + srow) * kE + schk * 8;

  v8f acc[2][4] = {};

#define STAGE_B(kci, buf) \
  async_load_b128((unsigned)(uintptr_t)&bsh[(buf)][srow][schk * 8], \
                  sbase + (kci) * 32)

  // Consume chunk kci from buffer buf; optionally issue nextk into nextbuf.
  auto step = [&](int kci, int buf, bool last, int nextk, int nextbuf) {
    if (last) wait_async0(); else wait_async1();   // own asyncs for kci done
    __syncthreads();                               // -> whole tile visible
    if (nextk >= 0) STAGE_B(nextk, nextbuf);       // buffer free since j-1
    const int kc = kci * 32;
    v16bf a0 = load_a_frag(hsb, kE, row0, kc, lane);
    v16bf a1 = load_a_frag(hsb, kE, row0 + 16, kc, lane);
    const __bf16* bbase = &bsh[buf][0][0];
#pragma unroll
    for (int t = 0; t < 4; ++t) {
      v16bf b = load_b_frag(bbase, 32, t * 16, 0, lane);  // ds_load_b128 x2
      acc[0][t] = wmma_bf16(a0, b, acc[0][t]);
      acc[1][t] = wmma_bf16(a1, b, acc[1][t]);
    }
  };

  STAGE_B(0, 0);
  STAGE_B(1, 1);
  for (int kci = 0; kci < 21; kci += 3) {          // chunks 0..20
    step(kci + 0, 0, false, kci + 2, 2);
    step(kci + 1, 1, false, kci + 3, 0);
    step(kci + 2, 2, false, kci + 4, 1);
  }
  step(21, 0, false, 23, 2);                       // tail: chunks 21..23
  step(22, 1, false, -1, 0);
  step(23, 2, true, -1, 0);
#undef STAGE_B

  const int hv = lane >> 4;
  const int nl = lane & 15;
#pragma unroll
  for (int t = 0; t < 4; ++t) {
    int e = n0 + t * 16 + nl;
    int hh = e >> 6;
    int d = e & 63;
    float bval = bias[e];
#pragma unroll
    for (int mt = 0; mt < 2; ++mt) {
#pragma unroll
      for (int r = 0; r < 8; ++r) {
        int m = row0 + mt * 16 + r + 8 * hv;  // global row
        int bb = m >> 12;                     // / 4096
        int s = m & 4095;
        float v = acc[mt][t][r] + bval;
        if (z == 0) {
          v *= 0.125f;  // 1/sqrt(64), reference pre-scales q
          qbuf[(((size_t)bb * kH + hh) * kS + s) * kD + d] = (__bf16)v;
        } else if (z == 1) {
          kbuf[(((size_t)bb * kH + hh) * kSP + (s + kW)) * kD + d] = (__bf16)v;
        } else {
          vtbuf[(((size_t)bb * kH + hh) * kD + d) * kSP + (s + kW)] = (__bf16)v;
        }
      }
    }
  }
}

// ---------------------------------------------------------------------------
// Kernel 3: sliding-window (banded) attention, flash-style streaming softmax.
// One wave per 16-query tile, 8 waves/block; 17 key blocks of 32 cover
// [q0-256, q0+287]; out-of-band / out-of-sequence positions masked to -1e9
// (self-correcting under the online rescale; matches the reference mask).
// ---------------------------------------------------------------------------
__global__ __launch_bounds__(256) void sliding_attn_kernel(
    const __bf16* __restrict__ qbuf, const __bf16* __restrict__ kbuf,
    const __bf16* __restrict__ vtbuf, float* __restrict__ out) {
  __shared__ __align__(16) __bf16 plds[8][16][32];  // per-wave P staging

  const int lane = threadIdx.x & 31;
  const int wave = threadIdx.x >> 5;
  const int tile = blockIdx.x * 8 + wave;  // 0..6143
  const int bh = tile >> 8;                // (b*H + h), S/16 == 256 tiles
  const int q0 = (tile & 255) << 4;
  const int bb = bh / kH;
  const int hh = bh % kH;

  const __bf16* Qp = qbuf + (size_t)bh * kS * kD;
  const __bf16* Kp = kbuf + (size_t)bh * kSP * kD;
  const __bf16* Vp = vtbuf + (size_t)bh * kD * kSP;

  // Q tile 16x64 held as two A fragments for the whole band.
  v16bf qa0 = load_a_frag(Qp, kD, q0, 0, lane);
  v16bf qa1 = load_a_frag(Qp, kD, q0, 32, lane);

  const int hv = lane >> 4;
  const int nl = lane & 15;

  float m[8], l[8];
  v8f o[4] = {};
#pragma unroll
  for (int r = 0; r < 8; ++r) { m[r] = -1.0e9f; l[r] = 0.0f; }

  for (int blk = 0; blk < 17; ++blk) {
    const int pk0 = q0 + blk * 32;  // padded key index of block start
    const int key0 = pk0 - kW;      // logical key index

    // --- issue all global loads for this iteration up front ---
    v16bf kb0a = load_b_frag(Kp, kD, pk0, 0, lane);
    v16bf kb0b = load_b_frag(Kp, kD, pk0, 32, lane);
    v16bf kb1a = load_b_frag(Kp, kD, pk0 + 16, 0, lane);
    v16bf kb1b = load_b_frag(Kp, kD, pk0 + 16, 32, lane);
    v16bf vb[4];
#pragma unroll
    for (int t = 0; t < 4; ++t) vb[t] = load_b_frag(Vp, kSP, t * 16, pk0, lane);

    // --- prefetch next key block (global_prefetch_b8) ---
    if (blk < 16) {
      const int pk1 = pk0 + 32;
      __builtin_prefetch(Kp + (size_t)(pk1 + lane) * kD, 0, 3);
      __builtin_prefetch(Vp + (size_t)lane * kSP + pk1, 0, 3);
      __builtin_prefetch(Vp + (size_t)(lane + 32) * kSP + pk1, 0, 3);
    }

    // --- S = Q K^T for 32 keys (two 16-key subtiles, K-dim 64 = 2x32) ---
    v8f s0 = {}, s1 = {};
    s0 = wmma_bf16(qa0, kb0a, s0);
    s0 = wmma_bf16(qa1, kb0b, s0);
    s1 = wmma_bf16(qa0, kb1a, s1);
    s1 = wmma_bf16(qa1, kb1b, s1);

    // --- mask + online softmax, per C-matrix row (r, half) ---
#pragma unroll
    for (int r = 0; r < 8; ++r) {
      const int qpos = q0 + r + 8 * hv;
      const int kA = key0 + nl;
      const int kBk = kA + 16;
      float e0 = s0[r];
      float e1 = s1[r];
      bool vA = (kA >= qpos - kW) && (kA <= qpos + kW) && (kA >= 0) && (kA < kS);
      bool vB =
          (kBk >= qpos - kW) && (kBk <= qpos + kW) && (kBk >= 0) && (kBk < kS);
      e0 = vA ? e0 : -1.0e9f;
      e1 = vB ? e1 : -1.0e9f;

      float tm = half_max(fmaxf(e0, e1));
      float mn = fmaxf(m[r], tm);
      float sc = __expf(m[r] - mn);
      float p0 = __expf(e0 - mn);
      float p1 = __expf(e1 - mn);
      float rs = half_sum(p0 + p1);
      l[r] = l[r] * sc + rs;
      m[r] = mn;
#pragma unroll
      for (int t = 0; t < 4; ++t) o[t][r] *= sc;

      // Stage P (C layout) to LDS row-major 16x32 for A-fragment reload.
      plds[wave][r + 8 * hv][nl] = (__bf16)p0;
      plds[wave][r + 8 * hv][nl + 16] = (__bf16)p1;
    }
    wave_lds_fence();

    // --- O += P V : P as A (16x32), V^T rows are contiguous B fragments ---
    v16bf pa = load_a_frag(&plds[wave][0][0], 32, 0, 0, lane);
#pragma unroll
    for (int t = 0; t < 4; ++t) o[t] = wmma_bf16(pa, vb[t], o[t]);
    wave_lds_fence();
  }

  // --- normalize and write (B, S, E) fp32 output ---
#pragma unroll
  for (int r = 0; r < 8; ++r) {
    float inv = 1.0f / l[r];
    int s = q0 + r + 8 * hv;
    float* op = out + ((size_t)bb * kS + s) * kE + hh * 64;
#pragma unroll
    for (int t = 0; t < 4; ++t) op[t * 16 + nl] = o[t][r] * inv;
  }
}

// ---------------------------------------------------------------------------
// Host launcher
// ---------------------------------------------------------------------------
extern "C" void kernel_launch(void* const* d_in, const int* in_sizes, int n_in,
                              void* d_out, int out_size, void* d_ws,
                              size_t ws_size, hipStream_t stream) {
  const float* hs = (const float*)d_in[0];
  // d_in[1] = attention_mask (all zeros in reference; unused)
  const float* Wq = (const float*)d_in[2];
  const float* bq = (const float*)d_in[3];
  const float* Wk = (const float*)d_in[4];
  const float* bk = (const float*)d_in[5];
  const float* Wv = (const float*)d_in[6];
  const float* bv = (const float*)d_in[7];
  float* out = (float*)d_out;

  // Workspace layout (bytes, all 256-aligned):
  constexpr size_t nHS = (size_t)kB * kS * kE;        // 6,291,456
  constexpr size_t nW = (size_t)kE * kE;              // 589,824
  constexpr size_t nQ = (size_t)kB * kH * kS * kD;    // 6,291,456
  constexpr size_t nKP = (size_t)kB * kH * kSP * kD;  // padded K
  char* ws = (char*)d_ws;
  __bf16* hsb = (__bf16*)(ws);
  __bf16* wall = (__bf16*)(ws + 2 * nHS);
  __bf16* qbuf = (__bf16*)(ws + 2 * (nHS + 3 * nW));
  __bf16* kbuf = (__bf16*)(ws + 2 * (nHS + 3 * nW + nQ));
  __bf16* vtbuf = (__bf16*)(ws + 2 * (nHS + 3 * nW + nQ + nKP));
  (void)ws_size; (void)in_sizes; (void)n_in; (void)out_size;

  // 1) fp32 -> bf16
  f32_to_bf16_kernel<<<(int)((nHS + 255) / 256), 256, 0, stream>>>(hs, hsb,
                                                                   (int)nHS);
  f32_to_bf16_kernel<<<(int)((nW + 255) / 256), 256, 0, stream>>>(Wq, wall,
                                                                  (int)nW);
  f32_to_bf16_kernel<<<(int)((nW + 255) / 256), 256, 0, stream>>>(
      Wk, wall + nW, (int)nW);
  f32_to_bf16_kernel<<<(int)((nW + 255) / 256), 256, 0, stream>>>(
      Wv, wall + 2 * nW, (int)nW);

  // 2) QKV projections: grid (M/256, E/64, {q,k,v}), 8 waves/block
  dim3 ggemm(kB * kS / 256, kE / 64, 3);
  qkv_gemm_kernel<<<ggemm, 256, 0, stream>>>(hsb, wall, bq, bk, bv, qbuf, kbuf,
                                             vtbuf);

  // 3) banded attention: (B*H*S/16) tiles, 8 waves/block
  int ntiles = kB * kH * (kS / 16);
  sliding_attn_kernel<<<ntiles / 8, 256, 0, stream>>>(qbuf, kbuf, vtbuf, out);
}
